// AMNTDDA_Fuzzy_1211180778231
// MI455X (gfx1250) — matrix operand order repaired and checked by reading, not compiled, not run on hardware
//
#include <hip/hip_runtime.h>

// ---------------------------------------------------------------------------
// Types for CDNA5 WMMA
// ---------------------------------------------------------------------------
typedef __attribute__((ext_vector_type(16))) __bf16 v16bf;
typedef __attribute__((ext_vector_type(8)))  float  v8f;

__device__ __forceinline__ unsigned short f2bfbits(float f) {
    union { float f; unsigned u; } a; a.f = f;
    unsigned u = a.u + 0x7FFFu + ((a.u >> 16) & 1u);   // round-to-nearest-even
    return (unsigned short)(u >> 16);
}

__device__ __forceinline__ v16bf load_frag2(const unsigned short* p0, const unsigned short* p1) {
    union { uint4 u[2]; v16bf v; } t;
    t.u[0] = *reinterpret_cast<const uint4*>(p0);
    t.u[1] = *reinterpret_cast<const uint4*>(p1);
    return t.v;
}

// ---------------------------------------------------------------------------
// Weight packing (B): f32 (K x N, row-major) -> bf16 fragments.
// Layout: dst[((k/16)*N + n)*16 + (k%16)], zero padded to Kp=ceil32(K).
// Each WMMA lane reads 16 consecutive bf16 (32B) for its column.
// ---------------------------------------------------------------------------
__global__ void pack_b(const float* __restrict__ src, unsigned short* __restrict__ dst,
                       int K, int N, int Kp, long sStride, long dStride) {
    src += (long)blockIdx.z * sStride;
    dst += (long)blockIdx.z * dStride;
    long idx = (long)blockIdx.x * blockDim.x + threadIdx.x;
    if (idx >= (long)Kp * N) return;
    int k = (int)(idx / N), n = (int)(idx % N);
    float v = (k < K) ? src[(size_t)k * N + n] : 0.f;
    dst[(((size_t)(k >> 4)) * N + n) * 16 + (k & 15)] = f2bfbits(v);
}

// ---------------------------------------------------------------------------
// Activation packing (A): f32 (M x K, leading dim lda) -> bf16 A-fragments.
// Layout: dst[((k/8)*M + r)*8 + (k%8)], zero padded to Kp=ceil32(K).
// Each WMMA lane then does two contiguous 16B loads per 32-wide K step.
// ---------------------------------------------------------------------------
__global__ void packa_k(const float* __restrict__ src, int lda, int M, int K, int Kp,
                        unsigned short* __restrict__ dst) {
    long idx = (long)blockIdx.x * blockDim.x + threadIdx.x;
    if (idx >= (long)M * Kp) return;
    int r = (int)(idx / Kp), k = (int)(idx % Kp);
    float v = (k < K) ? src[(size_t)r * lda + k] : 0.f;
    dst[(((size_t)(k >> 3)) * M + r) * 8 + (k & 7)] = f2bfbits(v);
}

// ---------------------------------------------------------------------------
// WMMA GEMM: C[z] = act(Apk[z] @ Bp[z] + bias + resid)
// A, B pre-packed bf16 (zero-padded K). Each wave computes a 16x32 tile
// (two accumulators sharing one A fragment -> halves A traffic), K-loop
// unrolled x2 => 4 v_wmma per iteration, branch-free.
// act: 0=none 1=relu 2=gelu(exact)
// ---------------------------------------------------------------------------
__global__ __launch_bounds__(128) void gemm_wmma(
    const unsigned short* __restrict__ Apk, long sAz,
    const unsigned short* __restrict__ Bp, long sBz,
    const float* __restrict__ bias,
    const float* __restrict__ resid, int ldr,
    float* __restrict__ C, int ldc, long sCz,
    int M, int N, int Kp, int act)
{
    int wave = threadIdx.x >> 5;
    int lane = threadIdx.x & 31;
    int n0 = (blockIdx.x * 4 + wave) * 32;     // 32 columns per wave
    if (n0 >= N) return;
    int m0 = blockIdx.y * 16;
    int z  = blockIdx.z;
    Apk += (long)z * sAz;
    Bp  += (long)z * sBz;
    C   += (long)z * sCz;

    bool hi  = lane >= 16;
    int  l15 = lane & 15;
    int  row  = m0 + l15;          // A row held by this lane
    int  col0 = n0 + l15;          // first B/C column held by this lane
    int  col1 = col0 + 16;         // second B/C column
    if (row >= M) row = M - 1;     // clamp (stores remain guarded)

    // per-lane base pointers
    const unsigned short* abase  = Apk + (size_t)row * 8;    // + kb * M*8
    const unsigned short* bbase0 = Bp + (size_t)col0 * 16;   // + k16 * N*16
    const unsigned short* bbase1 = Bp + (size_t)col1 * 16;
    size_t aStep = (size_t)M * 8;      // one k-block (8 values)
    size_t bStep = (size_t)N * 16;     // one k16-block (16 values)

    v8f acc0 = {0.f, 0.f, 0.f, 0.f, 0.f, 0.f, 0.f, 0.f};
    v8f acc1 = {0.f, 0.f, 0.f, 0.f, 0.f, 0.f, 0.f, 0.f};
    int nks = Kp >> 5;                 // 32-wide K steps
    int hib = hi ? 1 : 0;

    int ks = 0;
    for (; ks + 1 < nks; ks += 2) {
        int kb0 = (ks << 2) + 2 * hib;
        size_t bOff0 = (size_t)((ks << 1) + hib) * bStep;
        v16bf a0 = load_frag2(abase + (size_t)kb0 * aStep,
                              abase + (size_t)(kb0 + 1) * aStep);
        v16bf b00 = *reinterpret_cast<const v16bf*>(bbase0 + bOff0);
        v16bf b01 = *reinterpret_cast<const v16bf*>(bbase1 + bOff0);
        acc0 = __builtin_amdgcn_wmma_f32_16x16x32_bf16(false, a0, false, b00, (short)0, acc0, false, false);
        acc1 = __builtin_amdgcn_wmma_f32_16x16x32_bf16(false, a0, false, b01, (short)0, acc1, false, false);

        int kb1 = kb0 + 4;
        size_t bOff1 = (size_t)((ks << 1) + 2 + hib) * bStep;
        v16bf a1 = load_frag2(abase + (size_t)kb1 * aStep,
                              abase + (size_t)(kb1 + 1) * aStep);
        v16bf b10 = *reinterpret_cast<const v16bf*>(bbase0 + bOff1);
        v16bf b11 = *reinterpret_cast<const v16bf*>(bbase1 + bOff1);
        acc0 = __builtin_amdgcn_wmma_f32_16x16x32_bf16(false, a1, false, b10, (short)0, acc0, false, false);
        acc1 = __builtin_amdgcn_wmma_f32_16x16x32_bf16(false, a1, false, b11, (short)0, acc1, false, false);
    }
    if (ks < nks) {
        int kb0 = (ks << 2) + 2 * hib;
        size_t bOff0 = (size_t)((ks << 1) + hib) * bStep;
        v16bf a0 = load_frag2(abase + (size_t)kb0 * aStep,
                              abase + (size_t)(kb0 + 1) * aStep);
        v16bf b00 = *reinterpret_cast<const v16bf*>(bbase0 + bOff0);
        v16bf b01 = *reinterpret_cast<const v16bf*>(bbase1 + bOff0);
        acc0 = __builtin_amdgcn_wmma_f32_16x16x32_bf16(false, a0, false, b00, (short)0, acc0, false, false);
        acc1 = __builtin_amdgcn_wmma_f32_16x16x32_bf16(false, a0, false, b01, (short)0, acc1, false, false);
    }

    float bv0 = bias ? bias[col0] : 0.f;
    float bv1 = bias ? bias[col1] : 0.f;
    int rbase = m0 + (hi ? 8 : 0);
#pragma unroll
    for (int i = 0; i < 8; ++i) {
        int r = rbase + i;
        if (r < M) {
            float v0 = acc0[i] + bv0;
            float v1 = acc1[i] + bv1;
            if (resid) {
                v0 += resid[(size_t)r * ldr + col0];
                v1 += resid[(size_t)r * ldr + col1];
            }
            if (act == 1) {
                v0 = fmaxf(v0, 0.f);
                v1 = fmaxf(v1, 0.f);
            } else if (act == 2) {
                v0 = 0.5f * v0 * (1.f + erff(v0 * 0.70710678f));
                v1 = 0.5f * v1 * (1.f + erff(v1 * 0.70710678f));
            }
            C[(size_t)r * ldc + col0] = v0;
            C[(size_t)r * ldc + col1] = v1;
        }
    }
}

// ---------------------------------------------------------------------------
// Small helper kernels
// ---------------------------------------------------------------------------
__global__ void init_f(float* __restrict__ p, long n, float v) {
    long i = (long)blockIdx.x * blockDim.x + threadIdx.x;
    if (i < n) p[i] = v;
}

__global__ void gelu_k(float* __restrict__ p, long n) {
    long i = (long)blockIdx.x * blockDim.x + threadIdx.x;
    if (i >= n) return;
    float x = p[i];
    p[i] = 0.5f * x * (1.f + erff(x * 0.70710678f));
}

// LayerNorm over rows of 512, one wave per row
__global__ void ln512(const float* __restrict__ y, const float* __restrict__ g,
                      const float* __restrict__ b, float* __restrict__ out, int rows) {
    int wid  = (int)(((long)blockIdx.x * blockDim.x + threadIdx.x) >> 5);
    int lane = threadIdx.x & 31;
    if (wid >= rows) return;
    const float* yr = y + (size_t)wid * 512;
    float v[16];
    float s = 0.f;
#pragma unroll
    for (int i = 0; i < 16; ++i) { v[i] = yr[i * 32 + lane]; s += v[i]; }
#pragma unroll
    for (int o = 16; o > 0; o >>= 1) s += __shfl_xor(s, o, 32);
    float mean = s * (1.f / 512.f);
    float q = 0.f;
#pragma unroll
    for (int i = 0; i < 16; ++i) { float d = v[i] - mean; q += d * d; }
#pragma unroll
    for (int o = 16; o > 0; o >>= 1) q += __shfl_xor(q, o, 32);
    float r = rsqrtf(q * (1.f / 512.f) + 1e-5f);
    float* orow = out + (size_t)wid * 512;
#pragma unroll
    for (int i = 0; i < 16; ++i) {
        int c = i * 32 + lane;
        orow[c] = (v[i] - mean) * r * g[c] + b[c];
    }
}

// GT edge scores: sc[e,h] = dot(q[dst[e],h], k[src[e],h]) * scale
__global__ void edge_dot(const float* __restrict__ q, const float* __restrict__ k,
                         const int* __restrict__ src, const int* __restrict__ dst,
                         float* __restrict__ sc, int E, float scale) {
    long idx = (long)blockIdx.x * blockDim.x + threadIdx.x;
    if (idx >= (long)E * 8) return;
    int e = (int)(idx >> 3), h = (int)(idx & 7);
    const float* qr = q + (size_t)dst[e] * 512 + h * 64;
    const float* kr = k + (size_t)src[e] * 512 + h * 64;
    float s = 0.f;
#pragma unroll 8
    for (int d = 0; d < 64; ++d) s += qr[d] * kr[d];
    sc[idx] = s * scale;
}

// HGT edge scores: ke is per-edge-local (E x 512), scale *= pri[h]
__global__ void edge_dot_local(const float* __restrict__ ke, const float* __restrict__ q,
                               const int* __restrict__ dst, const float* __restrict__ prih,
                               float* __restrict__ sc, int E, float scale) {
    long idx = (long)blockIdx.x * blockDim.x + threadIdx.x;
    if (idx >= (long)E * 8) return;
    int e = (int)(idx >> 3), h = (int)(idx & 7);
    const float* kr = ke + (size_t)e * 512 + h * 64;
    const float* qr = q + (size_t)dst[e] * 512 + h * 64;
    float s = 0.f;
#pragma unroll 8
    for (int d = 0; d < 64; ++d) s += qr[d] * kr[d];
    sc[idx] = s * prih[h] * scale;
}

__device__ __forceinline__ void atomicMaxFloat(float* addr, float val) {
    if (val >= 0.f) atomicMax((int*)addr, __float_as_int(val));
    else            atomicMin((unsigned int*)addr, __float_as_uint(val));
}

__global__ void seg_max(const float* __restrict__ sc, const int* __restrict__ dst,
                        float* __restrict__ m, int E) {
    long idx = (long)blockIdx.x * blockDim.x + threadIdx.x;
    if (idx >= (long)E * 8) return;
    int e = (int)(idx >> 3), h = (int)(idx & 7);
    atomicMaxFloat(&m[(size_t)dst[e] * 8 + h], sc[idx]);
}

__global__ void seg_expsum(const float* __restrict__ sc, const int* __restrict__ dst,
                           const float* __restrict__ m, float* __restrict__ ev,
                           float* __restrict__ z, int E) {
    long idx = (long)blockIdx.x * blockDim.x + threadIdx.x;
    if (idx >= (long)E * 8) return;
    int e = (int)(idx >> 3), h = (int)(idx & 7);
    float eh = __expf(sc[idx] - m[(size_t)dst[e] * 8 + h]);
    ev[idx] = eh;
    atomicAdd(&z[(size_t)dst[e] * 8 + h], eh);
}

// agg[dst] += ev * v[src]  (v global, indexed by src)
__global__ void agg_gather(const float* __restrict__ ev, const int* __restrict__ src,
                           const int* __restrict__ dst, const float* __restrict__ v,
                           float* __restrict__ agg, int E) {
    long idx = (long)blockIdx.x * blockDim.x + threadIdx.x;
    if (idx >= (long)E * 8) return;
    int e = (int)(idx >> 3), h = (int)(idx & 7);
    float a = ev[idx];
    const float* vr = v + (size_t)src[e] * 512 + h * 64;
    float* ar = agg + (size_t)dst[e] * 512 + h * 64;
#pragma unroll 8
    for (int d = 0; d < 64; ++d) atomicAdd(&ar[d], a * vr[d]);
}

// agg[dst] += ev * me[e]  (me edge-local)
__global__ void agg_local(const float* __restrict__ ev, const int* __restrict__ dst,
                          const float* __restrict__ me, float* __restrict__ agg, int E) {
    long idx = (long)blockIdx.x * blockDim.x + threadIdx.x;
    if (idx >= (long)E * 8) return;
    int e = (int)(idx >> 3), h = (int)(idx & 7);
    float a = ev[idx];
    const float* vr = me + (size_t)e * 512 + h * 64;
    float* ar = agg + (size_t)dst[e] * 512 + h * 64;
#pragma unroll 8
    for (int d = 0; d < 64; ++d) atomicAdd(&ar[d], a * vr[d]);
}

__global__ void agg_norm(float* __restrict__ agg, const float* __restrict__ z, long total) {
    long idx = (long)blockIdx.x * blockDim.x + threadIdx.x;
    if (idx >= total) return;
    long row = idx >> 9;
    int h = (int)((idx >> 6) & 7);
    agg[idx] /= (z[row * 8 + h] + 1e-9f);
}

__global__ void gather_rows(const float* __restrict__ x, const int* __restrict__ idx,
                            float* __restrict__ out, int rows) {
    long i = (long)blockIdx.x * blockDim.x + threadIdx.x;
    if (i >= (long)rows * 512) return;
    long r = i >> 9, d = i & 511;
    out[i] = x[(size_t)idx[r] * 512 + d];
}

// x = sigmoid(skip[type]) * h + (1-sigmoid) * x ; node type from row ranges
__global__ void mix_skip(const float* __restrict__ h, const float* __restrict__ skip,
                         float* __restrict__ x, long total) {
    long i = (long)blockIdx.x * blockDim.x + threadIdx.x;
    if (i >= total) return;
    long row = i >> 9;
    int t = (row < 2048) ? 0 : (row < 3072 ? 1 : 2);
    float a = 1.f / (1.f + __expf(-skip[t]));
    x[i] = a * h[i] + (1.f - a) * x[i];
}

// out[(b,0,:)] = s0[b], out[(b,1,:)] = s1[b]
__global__ void interleave2(const float* __restrict__ s0, const float* __restrict__ s1,
                            float* __restrict__ out, int B) {
    long i = (long)blockIdx.x * blockDim.x + threadIdx.x;
    if (i >= (long)B * 512) return;
    long b = i >> 9, d = i & 511;
    out[(b * 2) * 512 + d]     = s0[i];
    out[(b * 2 + 1) * 512 + d] = s1[i];
}

// Full self-attention over S nodes, 2 channels, 8 heads, DH=64.
// Rows laid out (s*2 + c)*512 + h*64 + d. Online softmax; one wave per (s,c,h).
__global__ void attn_full(const float* __restrict__ q, const float* __restrict__ k,
                          const float* __restrict__ v, float* __restrict__ o, int S) {
    int gw   = (int)(((long)blockIdx.x * blockDim.x + threadIdx.x) >> 5);
    int lane = threadIdx.x & 31;
    if (gw >= S * 16) return;
    int s = gw >> 4;
    int c = (gw >> 3) & 1;
    int h = gw & 7;
    const float* qr = q + ((size_t)(s * 2 + c) * 512) + h * 64;
    float q0 = qr[lane], q1 = qr[lane + 32];
    float m = -3.4e38f, zz = 0.f, a0 = 0.f, a1 = 0.f;
    for (int t = 0; t < S; ++t) {
        size_t base = ((size_t)(t * 2 + c) * 512) + h * 64;
        const float* kr = k + base;
        float d = q0 * kr[lane] + q1 * kr[lane + 32];
#pragma unroll
        for (int x = 16; x > 0; x >>= 1) d += __shfl_xor(d, x, 32);
        d *= 0.125f;
        float mn   = fmaxf(m, d);
        float corr = __expf(m - mn);
        float e    = __expf(d - mn);
        const float* vr = v + base;
        zz = zz * corr + e;
        a0 = a0 * corr + e * vr[lane];
        a1 = a1 * corr + e * vr[lane + 32];
        m = mn;
    }
    float inv = 1.f / zz;
    float* orow = o + ((size_t)(s * 2 + c) * 512) + h * 64;
    orow[lane]      = a0 * inv;
    orow[lane + 32] = a1 * inv;
}

// emb[i] = dr[sample0]*di[sample1], 1024 wide, chunked
__global__ void emb_mul(const float* __restrict__ dr, const float* __restrict__ di,
                        const int* __restrict__ sample, float* __restrict__ out,
                        int n, int base) {
    long i = (long)blockIdx.x * blockDim.x + threadIdx.x;
    if (i >= (long)n * 1024) return;
    long r = i >> 10, cc = i & 1023;
    int s0 = sample[(base + r) * 2];
    int s1 = sample[(base + r) * 2 + 1];
    out[i] = dr[(size_t)s0 * 1024 + cc] * di[(size_t)s1 * 1024 + cc];
}

// Final 256 -> 2 layer (no activation)
__global__ void final2(const float* __restrict__ h, const float* __restrict__ w,
                       const float* __restrict__ b, float* __restrict__ out,
                       int n, int base) {
    int i = (int)((long)blockIdx.x * blockDim.x + threadIdx.x);
    if (i >= n) return;
    const float* hr = h + (size_t)i * 256;
    float a0 = b[0], a1 = b[1];
#pragma unroll 8
    for (int k = 0; k < 256; ++k) {
        float x = hr[k];
        a0 += x * w[k * 2];
        a1 += x * w[k * 2 + 1];
    }
    out[(size_t)(base + i) * 2]     = a0;
    out[(size_t)(base + i) * 2 + 1] = a1;
}

// ---------------------------------------------------------------------------
// Host orchestration
// ---------------------------------------------------------------------------
extern "C" void kernel_launch(void* const* d_in, const int* in_sizes, int n_in,
                              void* d_out, int out_size, void* d_ws, size_t ws_size,
                              hipStream_t stream)
{
    (void)in_sizes; (void)n_in; (void)out_size; (void)ws_size;
    const int DRUGN = 2048, DISN = 1024, NNODE = 5120;
    const int OFF1 = 2048, OFF2 = 3072;
    const int OFFS[3] = {0, OFF1, OFF2};
    const int SIZES[3] = {2048, 1024, 2048};

    const int* drdr      = (const int*)d_in[0];
    const int* didi      = (const int*)d_in[1];
    const int* het_src   = (const int*)d_in[2];
    const int* het_dst   = (const int*)d_in[3];
    const float* drug_feat = (const float*)d_in[6];
    const float* dis_feat  = (const float*)d_in[7];
    const float* prot_feat = (const float*)d_in[8];
    const int* sample    = (const int*)d_in[9];

    // ------------- param leaves (jax pytree order: dict keys sorted) -------
    int pi = 10;
    auto nxt = [&]() -> const float* { return (const float*)d_in[pi++]; };

    struct GTL { const float *b1,*b2,*ln1b,*ln1g,*ln2b,*ln2g,*w1,*w2,*wk,*wo,*wq,*wv; };
    struct HGTP { const float *A,*K,*Q,*V,*att,*msg,*pri,*skip; };
    struct ENC { const float *b1,*b2,*bk,*bo,*bq,*bv,*ln1b,*ln1g,*ln2b,*ln2g,*w1,*w2,*wk,*wo,*wq,*wv; };

    const float* drug_b = nxt(); const float* drug_w = nxt();

    const float* gtdis_emb = nxt();
    GTL gdis[2];
    for (int i = 0; i < 2; ++i) { GTL& L = gdis[i];
        L.b1=nxt(); L.b2=nxt(); L.ln1b=nxt(); L.ln1g=nxt(); L.ln2b=nxt(); L.ln2g=nxt();
        L.w1=nxt(); L.w2=nxt(); L.wk=nxt(); L.wo=nxt(); L.wq=nxt(); L.wv=nxt(); }

    const float* gtdrug_emb = nxt();
    GTL gdrug[2];
    for (int i = 0; i < 2; ++i) { GTL& L = gdrug[i];
        L.b1=nxt(); L.b2=nxt(); L.ln1b=nxt(); L.ln1g=nxt(); L.ln2b=nxt(); L.ln2g=nxt();
        L.w1=nxt(); L.w2=nxt(); L.wk=nxt(); L.wo=nxt(); L.wq=nxt(); L.wv=nxt(); }

    HGTP hg[2];
    for (int i = 0; i < 2; ++i) { HGTP& P = hg[i];
        P.A=nxt(); P.K=nxt(); P.Q=nxt(); P.V=nxt(); P.att=nxt(); P.msg=nxt(); P.pri=nxt(); P.skip=nxt(); }

    const float *mlp_b[4], *mlp_w[4];
    for (int i = 0; i < 4; ++i) { mlp_b[i]=nxt(); mlp_w[i]=nxt(); }

    const float* prot_b = nxt(); const float* prot_w = nxt();

    ENC edis[2];
    for (int i = 0; i < 2; ++i) { ENC& L = edis[i];
        L.b1=nxt(); L.b2=nxt(); L.bk=nxt(); L.bo=nxt(); L.bq=nxt(); L.bv=nxt();
        L.ln1b=nxt(); L.ln1g=nxt(); L.ln2b=nxt(); L.ln2g=nxt();
        L.w1=nxt(); L.w2=nxt(); L.wk=nxt(); L.wo=nxt(); L.wq=nxt(); L.wv=nxt(); }
    ENC edrug[2];
    for (int i = 0; i < 2; ++i) { ENC& L = edrug[i];
        L.b1=nxt(); L.b2=nxt(); L.bk=nxt(); L.bo=nxt(); L.bq=nxt(); L.bv=nxt();
        L.ln1b=nxt(); L.ln1g=nxt(); L.ln2b=nxt(); L.ln2g=nxt();
        L.w1=nxt(); L.w2=nxt(); L.wk=nxt(); L.wo=nxt(); L.wq=nxt(); L.wv=nxt(); }

    // ------------- workspace allocator -------------------------------------
    char* wp = (char*)d_ws;
    auto alloc = [&](size_t bytes) -> void* {
        void* p = (void*)wp;
        wp += (bytes + 255) & ~(size_t)255;
        return p;
    };
    auto nblk = [](long n) { return (unsigned)((n + 255) / 256); };

    // ------------- pack all weights to bf16 WMMA layout --------------------
    auto pack = [&](const float* src, int K, int N, int batch) -> const unsigned short* {
        int Kp = (K + 31) & ~31;
        unsigned short* dst = (unsigned short*)alloc((size_t)Kp * N * batch * 2);
        long per = (long)Kp * N;
        dim3 grid(nblk(per), 1, (unsigned)batch);
        pack_b<<<grid, 256, 0, stream>>>(src, dst, K, N, Kp, (long)K * N, per);
        return dst;
    };

    struct PGT { const unsigned short *w1,*w2,*wk,*wo,*wq,*wv; };
    struct PHGT { const unsigned short *A,*K,*Q,*V,*att,*msg; };
    struct PENC { const unsigned short *w1,*w2,*wk,*wo,*wq,*wv; };

    const unsigned short* pk_drug_w = pack(drug_w, 300, 512, 1);
    const unsigned short* pk_prot_w = pack(prot_w, 320, 512, 1);

    PGT pgdis[2], pgdrug[2];
    for (int i = 0; i < 2; ++i) {
        pgdis[i].w1 = pack(gdis[i].w1, 512, 1024, 1); pgdis[i].w2 = pack(gdis[i].w2, 1024, 512, 1);
        pgdis[i].wk = pack(gdis[i].wk, 512, 512, 1);  pgdis[i].wo = pack(gdis[i].wo, 512, 512, 1);
        pgdis[i].wq = pack(gdis[i].wq, 512, 512, 1);  pgdis[i].wv = pack(gdis[i].wv, 512, 512, 1);
        pgdrug[i].w1 = pack(gdrug[i].w1, 512, 1024, 1); pgdrug[i].w2 = pack(gdrug[i].w2, 1024, 512, 1);
        pgdrug[i].wk = pack(gdrug[i].wk, 512, 512, 1);  pgdrug[i].wo = pack(gdrug[i].wo, 512, 512, 1);
        pgdrug[i].wq = pack(gdrug[i].wq, 512, 512, 1);  pgdrug[i].wv = pack(gdrug[i].wv, 512, 512, 1);
    }
    PHGT phg[2];
    for (int i = 0; i < 2; ++i) {
        phg[i].A = pack(hg[i].A, 512, 512, 3); phg[i].K = pack(hg[i].K, 512, 512, 3);
        phg[i].Q = pack(hg[i].Q, 512, 512, 3); phg[i].V = pack(hg[i].V, 512, 512, 3);
        phg[i].att = pack(hg[i].att, 64, 64, 24); phg[i].msg = pack(hg[i].msg, 64, 64, 24);
    }
    PENC pedis[2], pedrug[2];
    for (int i = 0; i < 2; ++i) {
        pedis[i].w1 = pack(edis[i].w1, 512, 2048, 1); pedis[i].w2 = pack(edis[i].w2, 2048, 512, 1);
        pedis[i].wk = pack(edis[i].wk, 512, 512, 1);  pedis[i].wo = pack(edis[i].wo, 512, 512, 1);
        pedis[i].wq = pack(edis[i].wq, 512, 512, 1);  pedis[i].wv = pack(edis[i].wv, 512, 512, 1);
        pedrug[i].w1 = pack(edrug[i].w1, 512, 2048, 1); pedrug[i].w2 = pack(edrug[i].w2, 2048, 512, 1);
        pedrug[i].wk = pack(edrug[i].wk, 512, 512, 1);  pedrug[i].wo = pack(edrug[i].wo, 512, 512, 1);
        pedrug[i].wq = pack(edrug[i].wq, 512, 512, 1);  pedrug[i].wv = pack(edrug[i].wv, 512, 512, 1);
    }
    const unsigned short* pk_m0 = pack(mlp_w[0], 1024, 1024, 1);
    const unsigned short* pk_m1 = pack(mlp_w[1], 1024, 1024, 1);
    const unsigned short* pk_m2 = pack(mlp_w[2], 1024, 256, 1);

    // ------------- activation buffers --------------------------------------
    const size_t NODE = (size_t)NNODE * 512;
    float* xbuf   = (float*)alloc(NODE * 4);
    float* kbuf   = (float*)alloc(NODE * 4);
    float* qbuf   = (float*)alloc(NODE * 4);
    float* vbuf   = (float*)alloc(NODE * 4);
    float* aggbuf = (float*)alloc(NODE * 4);
    float* hbuf   = (float*)alloc(NODE * 4);
    float* ybuf   = (float*)alloc(NODE * 4);
    float* ffbuf  = (float*)alloc((size_t)4096 * 2048 * 4);
    float* g1     = (float*)alloc((size_t)16384 * 512 * 4);
    float* g2     = (float*)alloc((size_t)16384 * 512 * 4);
    float* sc_all = (float*)alloc((size_t)196608 * 8 * 4);
    float* ev_all = (float*)alloc((size_t)196608 * 8 * 4);
    float* mbuf   = (float*)alloc((size_t)NNODE * 8 * 4);
    float* zbuf   = (float*)alloc((size_t)NNODE * 8 * 4);
    float* dr_sim = (float*)alloc((size_t)DRUGN * 512 * 4);
    float* di_sim = (float*)alloc((size_t)DISN * 512 * 4);
    float* disq   = (float*)alloc((size_t)DISN * 1024 * 4);
    // packed-A scratch (largest: 16384x512 / 4096x2048 / 8192x1024 = 8.4M elems)
    unsigned short* paA = (unsigned short*)alloc((size_t)8388608 * 2);
    float* dout   = (float*)d_out;

    // pack activation matrix into paA
    auto packa = [&](const float* A, int lda, int M, int K) {
        int Kp = (K + 31) & ~31;
        packa_k<<<nblk((long)M * Kp), 256, 0, stream>>>(A, lda, M, K, Kp, paA);
    };

    auto gemm = [&](long sAz,
                    const unsigned short* Bp, long sBz,
                    const float* bias, const float* resid, int ldr,
                    float* C, int ldc, long sCz,
                    int M, int N, int Kp, int act, int nz) {
        int ntile = (N + 31) / 32;                    // 32 columns per wave
        dim3 grid((unsigned)((ntile + 3) / 4), (unsigned)((M + 15) / 16), (unsigned)nz);
        gemm_wmma<<<grid, 128, 0, stream>>>(paA, sAz, Bp, sBz, bias, resid, ldr,
                                            C, ldc, sCz, M, N, Kp, act);
    };

    // ------------- similarity graph transformers (_gt) ---------------------
    auto runGT = [&](const float* emb, int n, const int* src, const int* dst, int E,
                     GTL* Ls, PGT* Ps, float* out_sim) {
        hipMemcpyAsync(xbuf, emb, (size_t)n * 512 * 4, hipMemcpyDeviceToDevice, stream);
        for (int l = 0; l < 2; ++l) {
            GTL& L = Ls[l]; PGT& P = Ps[l];
            packa(xbuf, 512, n, 512);
            gemm(0, P.wq, 0, nullptr, nullptr, 0, qbuf, 512, 0, n, 512, 512, 0, 1);
            gemm(0, P.wk, 0, nullptr, nullptr, 0, kbuf, 512, 0, n, 512, 512, 0, 1);
            gemm(0, P.wv, 0, nullptr, nullptr, 0, vbuf, 512, 0, n, 512, 512, 0, 1);
            edge_dot<<<nblk((long)E * 8), 256, 0, stream>>>(qbuf, kbuf, src, dst, sc_all, E, 0.125f);
            init_f<<<nblk((long)n * 8), 256, 0, stream>>>(mbuf, (long)n * 8, -3.0e38f);
            init_f<<<nblk((long)n * 8), 256, 0, stream>>>(zbuf, (long)n * 8, 0.f);
            init_f<<<nblk((long)n * 512), 256, 0, stream>>>(aggbuf, (long)n * 512, 0.f);
            seg_max<<<nblk((long)E * 8), 256, 0, stream>>>(sc_all, dst, mbuf, E);
            seg_expsum<<<nblk((long)E * 8), 256, 0, stream>>>(sc_all, dst, mbuf, ev_all, zbuf, E);
            agg_gather<<<nblk((long)E * 8), 256, 0, stream>>>(ev_all, src, dst, vbuf, aggbuf, E);
            agg_norm<<<nblk((long)n * 512), 256, 0, stream>>>(aggbuf, zbuf, (long)n * 512);
            packa(aggbuf, 512, n, 512);
            gemm(0, P.wo, 0, nullptr, xbuf, 512, ybuf, 512, 0, n, 512, 512, 0, 1);
            ln512<<<nblk((long)n * 32), 256, 0, stream>>>(ybuf, L.ln1g, L.ln1b, xbuf, n);
            packa(xbuf, 512, n, 512);
            gemm(0, P.w1, 0, L.b1, nullptr, 0, ffbuf, 1024, 0, n, 1024, 512, 1, 1);
            packa(ffbuf, 1024, n, 1024);
            gemm(0, P.w2, 0, L.b2, xbuf, 512, ybuf, 512, 0, n, 512, 1024, 0, 1);
            ln512<<<nblk((long)n * 32), 256, 0, stream>>>(ybuf, L.ln2g, L.ln2b, xbuf, n);
        }
        hipMemcpyAsync(out_sim, xbuf, (size_t)n * 512 * 4, hipMemcpyDeviceToDevice, stream);
    };
    runGT(gtdrug_emb, DRUGN, drdr, drdr + 65536, 65536, gdrug, pgdrug, dr_sim);
    runGT(gtdis_emb, DISN, didi, didi + 32768, 32768, gdis, pgdis, di_sim);

    // ------------- HGT input projection ------------------------------------
    packa(drug_feat, 300, DRUGN, 300);
    gemm(0, pk_drug_w, 0, drug_b, nullptr, 0, xbuf, 512, 0, DRUGN, 512, 320, 0, 1);
    hipMemcpyAsync(xbuf + (size_t)OFF1 * 512, dis_feat, (size_t)DISN * 512 * 4,
                   hipMemcpyDeviceToDevice, stream);
    packa(prot_feat, 320, 2048, 320);
    gemm(0, pk_prot_w, 0, prot_b, nullptr, 0,
         xbuf + (size_t)OFF2 * 512, 512, 0, 2048, 512, 320, 0, 1);

    // ------------- HGT layers ----------------------------------------------
    const int CH = 16384;
    for (int l = 0; l < 2; ++l) {
        HGTP& P = hg[l]; PHGT& Q = phg[l];
        for (int t = 0; t < 3; ++t) {
            int M = SIZES[t];
            size_t off = (size_t)OFFS[t] * 512;
            packa(xbuf + off, 512, M, 512);
            gemm(0, Q.K + (size_t)t * 512 * 512, 0, nullptr, nullptr, 0, kbuf + off, 512, 0, M, 512, 512, 0, 1);
            gemm(0, Q.Q + (size_t)t * 512 * 512, 0, nullptr, nullptr, 0, qbuf + off, 512, 0, M, 512, 512, 0, 1);
            gemm(0, Q.V + (size_t)t * 512 * 512, 0, nullptr, nullptr, 0, vbuf + off, 512, 0, M, 512, 512, 0, 1);
        }
        init_f<<<nblk((long)NNODE * 8), 256, 0, stream>>>(mbuf, (long)NNODE * 8, -3.0e38f);
        init_f<<<nblk((long)NNODE * 8), 256, 0, stream>>>(zbuf, (long)NNODE * 8, 0.f);
        init_f<<<nblk((long)NNODE * 512), 256, 0, stream>>>(aggbuf, (long)NNODE * 512, 0.f);
        // pass A: attention scores (ke = gathered k @ att[t][h])
        for (int t = 0; t < 3; ++t)
            for (int c = 0; c < 4; ++c) {
                long eoff = (long)t * 65536 + (long)c * CH;
                const int* srcp = het_src + eoff;
                const int* dstp = het_dst + eoff;
                gather_rows<<<nblk((long)CH * 512), 256, 0, stream>>>(kbuf, srcp, g1, CH);
                packa(g1, 512, CH, 512);
                gemm((long)64 * CH, Q.att + (size_t)t * 8 * 4096, 4096, nullptr, nullptr, 0,
                     g2, 512, 64, CH, 64, 64, 0, 8);
                edge_dot_local<<<nblk((long)CH * 8), 256, 0, stream>>>(
                    g2, qbuf, dstp, P.pri + t * 8, sc_all + eoff * 8, CH, 0.125f);
            }
        seg_max<<<nblk((long)196608 * 8), 256, 0, stream>>>(sc_all, het_dst, mbuf, 196608);
        seg_expsum<<<nblk((long)196608 * 8), 256, 0, stream>>>(sc_all, het_dst, mbuf, ev_all, zbuf, 196608);
        // pass B: messages (me = gathered v @ msg[t][h]) + aggregation
        for (int t = 0; t < 3; ++t)
            for (int c = 0; c < 4; ++c) {
                long eoff = (long)t * 65536 + (long)c * CH;
                const int* srcp = het_src + eoff;
                const int* dstp = het_dst + eoff;
                gather_rows<<<nblk((long)CH * 512), 256, 0, stream>>>(vbuf, srcp, g1, CH);
                packa(g1, 512, CH, 512);
                gemm((long)64 * CH, Q.msg + (size_t)t * 8 * 4096, 4096, nullptr, nullptr, 0,
                     g2, 512, 64, CH, 64, 64, 0, 8);
                agg_local<<<nblk((long)CH * 8), 256, 0, stream>>>(
                    ev_all + eoff * 8, dstp, g2, aggbuf, CH);
            }
        agg_norm<<<nblk((long)NNODE * 512), 256, 0, stream>>>(aggbuf, zbuf, (long)NNODE * 512);
        gelu_k<<<nblk((long)NNODE * 512), 256, 0, stream>>>(aggbuf, (long)NNODE * 512);
        for (int t = 0; t < 3; ++t) {
            int M = SIZES[t];
            size_t off = (size_t)OFFS[t] * 512;
            packa(aggbuf + off, 512, M, 512);
            gemm(0, Q.A + (size_t)t * 512 * 512, 0, nullptr, nullptr, 0, hbuf + off, 512, 0, M, 512, 512, 0, 1);
        }
        mix_skip<<<nblk((long)NNODE * 512), 256, 0, stream>>>(hbuf, P.skip, xbuf, (long)NNODE * 512);
    }

    // ------------- stack (sim, hgt) and transformer encoders ---------------
    interleave2<<<nblk((long)DRUGN * 512), 256, 0, stream>>>(dr_sim, xbuf, dout, DRUGN);
    interleave2<<<nblk((long)DISN * 512), 256, 0, stream>>>(di_sim, xbuf + (size_t)OFF1 * 512, disq, DISN);

    auto runEnc = [&](float* xs, int S, ENC* Ls, PENC* Ps) {
        int R = S * 2;
        for (int l = 0; l < 2; ++l) {
            ENC& L = Ls[l]; PENC& P = Ps[l];
            packa(xs, 512, R, 512);
            gemm(0, P.wq, 0, L.bq, nullptr, 0, qbuf, 512, 0, R, 512, 512, 0, 1);
            gemm(0, P.wk, 0, L.bk, nullptr, 0, kbuf, 512, 0, R, 512, 512, 0, 1);
            gemm(0, P.wv, 0, L.bv, nullptr, 0, vbuf, 512, 0, R, 512, 512, 0, 1);
            attn_full<<<nblk((long)S * 16 * 32), 256, 0, stream>>>(qbuf, kbuf, vbuf, aggbuf, S);
            packa(aggbuf, 512, R, 512);
            gemm(0, P.wo, 0, L.bo, xs, 512, ybuf, 512, 0, R, 512, 512, 0, 1);
            ln512<<<nblk((long)R * 32), 256, 0, stream>>>(ybuf, L.ln1g, L.ln1b, xs, R);
            packa(xs, 512, R, 512);
            gemm(0, P.w1, 0, L.b1, nullptr, 0, ffbuf, 2048, 0, R, 2048, 512, 1, 1);
            packa(ffbuf, 2048, R, 2048);
            gemm(0, P.w2, 0, L.b2, xs, 512, ybuf, 512, 0, R, 512, 2048, 0, 1);
            ln512<<<nblk((long)R * 32), 256, 0, stream>>>(ybuf, L.ln2g, L.ln2b, xs, R);
        }
    };
    runEnc(dout, DRUGN, edrug, pedrug);   // dr -> d_out[0 .. 2048*1024)
    runEnc(disq, DISN, edis, pedis);

    // ------------- sampled pair MLP (chunked) ------------------------------
    float* hout = dout + (size_t)DRUGN * 1024;
    for (int c = 0; c < 8; ++c) {
        int base = c * 8192;
        emb_mul<<<nblk((long)8192 * 1024), 256, 0, stream>>>(dout, disq, sample, g1, 8192, base);
        packa(g1, 1024, 8192, 1024);
        gemm(0, pk_m0, 0, mlp_b[0], nullptr, 0, g2, 1024, 0, 8192, 1024, 1024, 1, 1);
        packa(g2, 1024, 8192, 1024);
        gemm(0, pk_m1, 0, mlp_b[1], nullptr, 0, g1, 1024, 0, 8192, 1024, 1024, 1, 1);
        packa(g1, 1024, 8192, 1024);
        gemm(0, pk_m2, 0, mlp_b[2], nullptr, 0, aggbuf, 256, 0, 8192, 256, 1024, 1, 1);
        final2<<<nblk(8192), 256, 0, stream>>>(aggbuf, mlp_w[3], mlp_b[3], hout, 8192, base);
    }
}
